// MultiHeadAttention_2087354106435
// MI455X (gfx1250) — matrix operand order repaired and checked by reading, not compiled, not run on hardware
//
#include <hip/hip_runtime.h>
#include <hip/hip_bf16.h>

// ---------------------------------------------------------------- constants
#define B_   2
#define N_   2048
#define D_   1024
#define H_   16
#define DH_  64
#define M_   (B_ * N_)          // 4096 rows for the big GEMMs

typedef __bf16 bf16;
typedef __attribute__((ext_vector_type(16))) __bf16 v16bf;
typedef __attribute__((ext_vector_type(8)))  __bf16 v8bf;
typedef __attribute__((ext_vector_type(8)))  float  v8f;

// s_wait_asynccnt <= n  (SOPP opcode 74) — inline asm avoids builtin-signature
// differences between toolchains.
#define WAIT_ASYNC(n) asm volatile("s_wait_asynccnt %0" ::"n"(n) : "memory")

// Async global->LDS 16B copy (ASYNCcnt-tracked).  VDST = LDS byte address
// (flat pointer truncated to 32 bits == LDS offset per ISA 10.2 aperture calc),
// VADDR = 64-bit global address.
__device__ __forceinline__ void async_copy_b128(const void* gsrc, void* ldst) {
    asm volatile("global_load_async_to_lds_b128 %0, %1, off"
                 :
                 : "v"((unsigned)(unsigned long long)ldst), "v"(gsrc)
                 : "memory");
}

__device__ __forceinline__ v8f zero_v8f() {
    v8f z;
#pragma unroll
    for (int i = 0; i < 8; ++i) z[i] = 0.0f;
    return z;
}

__device__ __forceinline__ v8f wmma_bf16(v16bf a, v16bf b, v8f c) {
    // D = A(16x32 bf16) * B(32x16 bf16) + C(16x16 f32)
    return __builtin_amdgcn_wmma_f32_16x16x32_bf16(
        /*neg_a=*/false, a, /*neg_b=*/false, b,
        /*c_mod=*/(short)0, c, /*reuse_a=*/false, /*reuse_b=*/false);
}

// A-fragment (16x32, bf16, row-major source, K contiguous).
// Layout: lanes 0-15 -> M=lane, elems 0-15 = K[0..15]
//         lanes 16-31 -> M=lane-16, elems 0-15 = K[16..31]
__device__ __forceinline__ v16bf load_a_frag(const bf16* __restrict__ base,
                                             int ld, int row, int k0, int lane) {
    const bf16* p = base + (size_t)row * ld + k0 + ((lane >> 4) << 4);
    v8bf lo = *(const v8bf*)(p);
    v8bf hi = *(const v8bf*)(p + 8);
    v16bf a;
#pragma unroll
    for (int i = 0; i < 8; ++i) { a[i] = lo[i]; a[i + 8] = hi[i]; }
    return a;
}

// B-fragment (32x16, bf16) where B[k][col] = base[col*ld + k] (K contiguous
// per column). Contiguous k order: lanes 0-15 -> col=lane, K[0..15]; lanes 16-31 -> K[16..31]
__device__ __forceinline__ v16bf load_b_frag(const bf16* __restrict__ base,
                                             int ld, int col0, int k0, int lane) {
    const bf16* p = base + (size_t)(col0 + (lane & 15)) * ld + k0 + ((lane >> 4) << 4);
    return *(const v16bf*)(p);
}

// Stage a 64-col x 32-K bf16 W tile into LDS ([col][k], 4KB) with 128 threads.
// Two async 16B copies per thread -> ASYNCcnt == 2 per stage per wave.
__device__ __forceinline__ void stage_w_tile(const bf16* __restrict__ W,
                                             int col0, int k0,
                                             bf16* lds_buf, int tid) {
#pragma unroll
    for (int i = 0; i < 2; ++i) {
        const int idx = tid + i * 128;       // 256 x 16B segments
        const int col = idx >> 2;            // 0..63
        const int seg = idx & 3;             // 16B segment of the 64B row
        const bf16* src = W + (size_t)(col0 + col) * D_ + k0 + seg * 8;
        bf16* dst = lds_buf + col * 32 + seg * 8;
        async_copy_b128(src, dst);
    }
}

// ---------------------------------------------------------------- cast
__global__ void cast_f32_bf16(const float* __restrict__ in, bf16* __restrict__ out, int n) {
    int i = blockIdx.x * 256 + threadIdx.x;
    if (i < n) out[i] = (bf16)in[i];
}

// ---------------------------------------------------------------- Q/K/V projections
// 4 waves / block; block tile 128x64; W tile async-staged + double-buffered in
// LDS and shared by all waves. Q,K stored [b,h,n,dh]; V stored transposed
// [b,h,dh,n] so the P*V operand in attention is contiguous.
__global__ void __launch_bounds__(128)
proj_gemm_kernel(const bf16* __restrict__ X,
                 const bf16* __restrict__ Wq,
                 const bf16* __restrict__ Wk,
                 const bf16* __restrict__ Wv,
                 bf16* __restrict__ Qo,
                 bf16* __restrict__ Ko,
                 bf16* __restrict__ Vt) {
    __shared__ alignas(64) bf16 sW[2][64 * 32];

    const int z = blockIdx.z;
    const bf16* W = (z == 0) ? Wq : (z == 1) ? Wk : Wv;

    const int tid  = threadIdx.x;
    const int lane = tid & 31;
    const int wv   = tid >> 5;
    const int hi   = lane >> 4;
    const int row0 = blockIdx.x * 128 + wv * 32;
    const int col0 = blockIdx.y * 64;

    v8f acc[2][4];
#pragma unroll
    for (int u = 0; u < 2; ++u)
#pragma unroll
        for (int c = 0; c < 4; ++c) acc[u][c] = zero_v8f();

    stage_w_tile(W, col0, 0, sW[0], tid);

    const int NSTEP = D_ / 32;
    for (int step = 0; step < NSTEP; ++step) {
        const int cur = step & 1;
        if (step + 1 < NSTEP) {
            stage_w_tile(W, col0, (step + 1) * 32, sW[cur ^ 1], tid);
            WAIT_ASYNC(2);     // current buffer landed; next stage in flight
        } else {
            WAIT_ASYNC(0);
        }
        __syncthreads();

        const int k0 = step * 32;
        v16bf a0 = load_a_frag(X, D_, row0 + (lane & 15),      k0, lane);
        v16bf a1 = load_a_frag(X, D_, row0 + 16 + (lane & 15), k0, lane);
#pragma unroll
        for (int c = 0; c < 4; ++c) {
            const bf16* bp = &sW[cur][(c * 16 + (lane & 15)) * 32 + (hi << 4)];
            v16bf b = *(const v16bf*)bp;
            acc[0][c] = wmma_bf16(a0, b, acc[0][c]);
            acc[1][c] = wmma_bf16(a1, b, acc[1][c]);
        }
        __syncthreads();       // protect buffer against next stage's overwrite
    }

#pragma unroll
    for (int c = 0; c < 4; ++c) {
        const int e  = col0 + c * 16 + (lane & 15);
        const int h  = e >> 6;
        const int dh = e & 63;
#pragma unroll
        for (int u = 0; u < 2; ++u) {
#pragma unroll
            for (int r = 0; r < 8; ++r) {
                const int m  = row0 + u * 16 + r + (hi << 3);
                const int b  = m >> 11;          // / N_
                const int nq = m & (N_ - 1);
                const bf16 val = (bf16)acc[u][c][r];
                if (z == 2) {
                    Vt[(((size_t)b * H_ + h) * DH_ + dh) * N_ + nq] = val;   // [b,h,dh,n]
                } else {
                    bf16* Ot = (z == 0) ? Qo : Ko;
                    Ot[(((size_t)b * H_ + h) * N_ + nq) * DH_ + dh] = val;   // [b,h,n,dh]
                }
            }
        }
    }
}

// ---------------------------------------------------------------- flash attention
// One wave handles a 16-query tile of one (b,h). Streams 32 keys / iteration.
// V is consumed pre-transposed: Vt[b,h,dh,n] -> contiguous B-fragment loads.
__global__ void attn_kernel(const bf16* __restrict__ Q,
                            const bf16* __restrict__ K,
                            const bf16* __restrict__ Vt,
                            bf16* __restrict__ Ctx) {
    __shared__ alignas(32) bf16 pstage[16 * 32];

    const int lane = threadIdx.x;
    const int hi   = lane >> 4;
    const int qt   = blockIdx.x;   // 0 .. N/16-1
    const int h    = blockIdx.y;
    const int b    = blockIdx.z;

    const size_t headOff = ((size_t)b * H_ + h) * N_ * DH_;
    const bf16* Qh  = Q  + headOff;
    const bf16* Kh  = K  + headOff;
    const bf16* Vth = Vt + headOff;   // [dh][n], ld = N_

    const int rowA = qt * 16 + (lane & 15);
    const v16bf aQ0 = load_a_frag(Qh, DH_, rowA, 0,  lane);
    const v16bf aQ1 = load_a_frag(Qh, DH_, rowA, 32, lane);

    float m[8], l[8];
    v8f O[4];
#pragma unroll
    for (int r = 0; r < 8; ++r) { m[r] = -3.0e38f; l[r] = 0.0f; }
#pragma unroll
    for (int c = 0; c < 4; ++c) O[c] = zero_v8f();

    const float scale = 0.02209708691207961f;  // 1/sqrt(N=2048)
    const int nkb = (qt >> 1) + 1;             // causal: only key blocks <= q

    for (int kb = 0; kb < nkb; ++kb) {
        const int j0 = kb * 32;
        // ---- S = Q * K^T for 32 keys (two 16-wide C tiles) --------------
        v8f S0 = zero_v8f(), S1 = zero_v8f();
        {
            v16bf b00 = load_b_frag(Kh, DH_, j0,      0,  lane);
            v16bf b01 = load_b_frag(Kh, DH_, j0,      32, lane);
            v16bf b10 = load_b_frag(Kh, DH_, j0 + 16, 0,  lane);
            v16bf b11 = load_b_frag(Kh, DH_, j0 + 16, 32, lane);
            S0 = wmma_bf16(aQ0, b00, S0);
            S0 = wmma_bf16(aQ1, b01, S0);
            S1 = wmma_bf16(aQ0, b10, S1);
            S1 = wmma_bf16(aQ1, b11, S1);
        }
        // ---- scale + causal mask + online softmax (in C-frag layout) ----
        const int col0 = j0 + (lane & 15);
        float p0[8], p1[8], corr[8];
#pragma unroll
        for (int r = 0; r < 8; ++r) {
            const int qrow = qt * 16 + r + (hi << 3);
            float s0 = S0[r] * scale;
            float s1 = S1[r] * scale;
            if (col0      > qrow) s0 = -3.0e38f;
            if (col0 + 16 > qrow) s1 = -3.0e38f;
            float mt = fmaxf(s0, s1);
            mt = fmaxf(mt, __shfl_xor(mt, 1, 32));
            mt = fmaxf(mt, __shfl_xor(mt, 2, 32));
            mt = fmaxf(mt, __shfl_xor(mt, 4, 32));
            mt = fmaxf(mt, __shfl_xor(mt, 8, 32));
            const float mnew = fmaxf(m[r], mt);
            corr[r] = __expf(m[r] - mnew);
            p0[r] = __expf(s0 - mnew);
            p1[r] = __expf(s1 - mnew);
            float ps = p0[r] + p1[r];
            ps += __shfl_xor(ps, 1, 32);
            ps += __shfl_xor(ps, 2, 32);
            ps += __shfl_xor(ps, 4, 32);
            ps += __shfl_xor(ps, 8, 32);
            l[r] = l[r] * corr[r] + ps;
            m[r] = mnew;
        }
#pragma unroll
        for (int c = 0; c < 4; ++c)
#pragma unroll
            for (int r = 0; r < 8; ++r) O[c][r] *= corr[r];

        // ---- P: C-frag -> A-frag via LDS --------------------------------
        __syncthreads();
#pragma unroll
        for (int r = 0; r < 8; ++r) {
            const int prow = r + (hi << 3);
            pstage[prow * 32 + (lane & 15)]      = (bf16)p0[r];
            pstage[prow * 32 + (lane & 15) + 16] = (bf16)p1[r];
        }
        __syncthreads();
        {
            const bf16* pp = &pstage[(lane & 15) * 32 + (hi << 4)];
            v8bf lo = *(const v8bf*)(pp);
            v8bf hc = *(const v8bf*)(pp + 8);
            v16bf aP;
#pragma unroll
            for (int i = 0; i < 8; ++i) { aP[i] = lo[i]; aP[i + 8] = hc[i]; }
            // ---- O += P * V  (Vt: col=dh, K=key contiguous) -------------
#pragma unroll
            for (int c = 0; c < 4; ++c) {
                v16bf bV = load_b_frag(Vth, N_, c * 16, j0, lane);
                O[c] = wmma_bf16(aP, bV, O[c]);
            }
        }
    }

    // ---- normalize and write ctx[b, n, h*64+dh] as bf16 -----------------
#pragma unroll
    for (int r = 0; r < 8; ++r) {
        const float inv  = 1.0f / l[r];
        const int   qrow = qt * 16 + r + (hi << 3);
#pragma unroll
        for (int c = 0; c < 4; ++c) {
            const int d = h * 64 + c * 16 + (lane & 15);
            Ctx[((size_t)b * N_ + qrow) * D_ + d] = (bf16)(O[c][r] * inv);
        }
    }
}

// ---------------------------------------------------------------- out = ctx*Wo^T + bo
// Same block structure as proj: 4 waves, async-staged double-buffered Wo tile.
__global__ void __launch_bounds__(128)
out_gemm_kernel(const bf16* __restrict__ CtxB,
                const bf16* __restrict__ WoB,
                const float* __restrict__ bo,
                float* __restrict__ Out) {
    __shared__ alignas(64) bf16 sW[2][64 * 32];

    const int tid  = threadIdx.x;
    const int lane = tid & 31;
    const int wv   = tid >> 5;
    const int hi   = lane >> 4;
    const int row0 = blockIdx.x * 128 + wv * 32;
    const int col0 = blockIdx.y * 64;

    v8f acc[2][4];
#pragma unroll
    for (int u = 0; u < 2; ++u)
#pragma unroll
        for (int c = 0; c < 4; ++c) acc[u][c] = zero_v8f();

    stage_w_tile(WoB, col0, 0, sW[0], tid);

    const int NSTEP = D_ / 32;
    for (int step = 0; step < NSTEP; ++step) {
        const int cur = step & 1;
        if (step + 1 < NSTEP) {
            stage_w_tile(WoB, col0, (step + 1) * 32, sW[cur ^ 1], tid);
            WAIT_ASYNC(2);
        } else {
            WAIT_ASYNC(0);
        }
        __syncthreads();

        const int k0 = step * 32;
        v16bf a0 = load_a_frag(CtxB, D_, row0 + (lane & 15),      k0, lane);
        v16bf a1 = load_a_frag(CtxB, D_, row0 + 16 + (lane & 15), k0, lane);
#pragma unroll
        for (int c = 0; c < 4; ++c) {
            const bf16* bp = &sW[cur][(c * 16 + (lane & 15)) * 32 + (hi << 4)];
            v16bf b = *(const v16bf*)bp;
            acc[0][c] = wmma_bf16(a0, b, acc[0][c]);
            acc[1][c] = wmma_bf16(a1, b, acc[1][c]);
        }
        __syncthreads();
    }

#pragma unroll
    for (int c = 0; c < 4; ++c) {
        const int   e  = col0 + c * 16 + (lane & 15);
        const float bb = bo[e];
#pragma unroll
        for (int u = 0; u < 2; ++u) {
#pragma unroll
            for (int r = 0; r < 8; ++r) {
                const int mrow = row0 + u * 16 + r + (hi << 3);
                Out[(size_t)mrow * D_ + e] = acc[u][c][r] + bb;
            }
        }
    }
}

// ---------------------------------------------------------------- launch
extern "C" void kernel_launch(void* const* d_in, const int* in_sizes, int n_in,
                              void* d_out, int out_size, void* d_ws, size_t ws_size,
                              hipStream_t stream) {
    const float* x  = (const float*)d_in[0];
    const float* Wq = (const float*)d_in[1];
    const float* Wk = (const float*)d_in[2];
    const float* Wv = (const float*)d_in[3];
    const float* Wo = (const float*)d_in[4];
    const float* bo = (const float*)d_in[5];
    float* out = (float*)d_out;

    char*  ws  = (char*)d_ws;
    size_t off = 0;
    auto alloc = [&](size_t bytes) -> char* {
        char* p = ws + off;
        off += (bytes + 255) & ~(size_t)255;
        return p;
    };
    bf16* xb  = (bf16*)alloc((size_t)M_ * D_ * sizeof(bf16));
    bf16* wqb = (bf16*)alloc((size_t)D_ * D_ * sizeof(bf16));
    bf16* wkb = (bf16*)alloc((size_t)D_ * D_ * sizeof(bf16));
    bf16* wvb = (bf16*)alloc((size_t)D_ * D_ * sizeof(bf16));
    bf16* wob = (bf16*)alloc((size_t)D_ * D_ * sizeof(bf16));
    bf16* Qb  = (bf16*)alloc((size_t)M_ * D_ * sizeof(bf16));
    bf16* Kb  = (bf16*)alloc((size_t)M_ * D_ * sizeof(bf16));
    bf16* Vtb = (bf16*)alloc((size_t)M_ * D_ * sizeof(bf16));
    bf16* Cb  = (bf16*)alloc((size_t)M_ * D_ * sizeof(bf16));

    const int nX = M_ * D_;
    const int nW = D_ * D_;
    cast_f32_bf16<<<(nX + 255) / 256, 256, 0, stream>>>(x,  xb,  nX);
    cast_f32_bf16<<<(nW + 255) / 256, 256, 0, stream>>>(Wq, wqb, nW);
    cast_f32_bf16<<<(nW + 255) / 256, 256, 0, stream>>>(Wk, wkb, nW);
    cast_f32_bf16<<<(nW + 255) / 256, 256, 0, stream>>>(Wv, wvb, nW);
    cast_f32_bf16<<<(nW + 255) / 256, 256, 0, stream>>>(Wo, wob, nW);

    dim3 gProj(M_ / 128, D_ / 64, 3);
    proj_gemm_kernel<<<gProj, 128, 0, stream>>>(xb, wqb, wkb, wvb, Qb, Kb, Vtb);

    dim3 gAttn(N_ / 16, H_, B_);
    attn_kernel<<<gAttn, 32, 0, stream>>>(Qb, Kb, Vtb, Cb);

    dim3 gOut(M_ / 128, D_ / 64);
    out_gemm_kernel<<<gOut, 128, 0, stream>>>(Cb, wob, bo, out);
}
